// Block_43344809951338
// MI455X (gfx1250) — matrix operand order repaired and checked by reading, not compiled, not run on hardware
//
#include <hip/hip_runtime.h>

// ---------------------------------------------------------------------------
// Transformer encoder block (GPT-2 small geometry) for MI455X / gfx1250.
// GEMMs + attention on v_wmma_f32_16x16x32_bf16; GEMM tile staging via the
// Tensor Data Mover (tensor_load_to_lds + s_wait_tensorcnt) when available.
// ---------------------------------------------------------------------------

typedef __attribute__((ext_vector_type(16))) __bf16 v16bf;
typedef __attribute__((ext_vector_type(8)))  float  v8f;
typedef unsigned int   u32x4 __attribute__((ext_vector_type(4)));
typedef int            i32x8 __attribute__((ext_vector_type(8)));
typedef int            i32x4 __attribute__((ext_vector_type(4)));
typedef unsigned short u16;
typedef unsigned int   u32;
typedef unsigned long long u64;

#define EMBED   768
#define SEQ     1024
#define BATCH   8
#define TOKENS  (BATCH * SEQ)        // 8192
#define HEADS   12
#define HDIM    64
#define HIDDEN  (4 * EMBED)          // 3072
#define QKVN    (3 * EMBED)          // 2304

#if __has_builtin(__builtin_amdgcn_tensor_load_to_lds) && \
    __has_builtin(__builtin_amdgcn_s_wait_tensorcnt)
#define USE_TDM 1
#endif

__device__ __forceinline__ u16 f2bf(float f) {
  union { float f; u32 u; } v; v.f = f;
  u32 r = v.u + 0x7FFFu + ((v.u >> 16) & 1u);   // round-to-nearest-even
  return (u16)(r >> 16);
}

__device__ __forceinline__ v8f v8f_zero() {
  v8f z;
#pragma unroll
  for (int e = 0; e < 8; ++e) z[e] = 0.0f;
  return z;
}

union Frag { uint4 q[2]; v16bf v; };

#ifdef USE_TDM
// Issue one TDM load of a 2D bf16 tile: 128 rows x 32 elements, row stride
// `rowlen` elements, into LDS at lds_off with 16B padding after each 64B row
// (pad_interval = 16 DWORDs, pad_amount = 4 DWORDs -> 80B LDS row stride).
// Descriptor layout per CDNA5 ISA section 8 (D# groups 0/1).
// This toolchain's builtin takes 6 args: (g0, g1, g2, g3, g4, cpol).
__device__ __forceinline__ void tdm_load_tile(const u16* gbase, u32 lds_off,
                                              int rowlen, int rows) {
  const u64 ga = (u64)(size_t)gbase;
  u32x4 g0;
  g0[0] = 1u;                                   // count=1, user descriptor
  g0[1] = lds_off;                              // lds_addr
  g0[2] = (u32)ga;                              // global_addr[31:0]
  g0[3] = (u32)(ga >> 32) | (2u << 30);         // global_addr[56:32] | type=2
  i32x8 g1;
  g1[0] = (1 << 16) | (1 << 20) | (3 << 22) | (3 << 25);
         // data_size=2B | pad_enable | pad_interval=16DW | pad_amount=4DW
  g1[1] = (rowlen & 0xFFFF) << 16;                              // tensor_dim0 lo
  g1[2] = ((rowlen >> 16) & 0xFFFF) | ((rows & 0xFFFF) << 16);  // dim0 hi|dim1 lo
  g1[3] = ((rows >> 16) & 0xFFFF) | (32 << 16);                 // dim1 hi|tile_dim0=32
  g1[4] = 128;                                                  // tile_dim1=128
  g1[5] = rowlen;                                               // dim0_stride lo32
  g1[6] = 0;
  g1[7] = 0;
  const i32x4 z4 = {0, 0, 0, 0};
  const i32x8 z8 = {0, 0, 0, 0, 0, 0, 0, 0};
  __builtin_amdgcn_tensor_load_to_lds(g0, g1, z4, z4, z8, 0);
}
#endif

// ---------------------------------------------------------------------------
// fp32 [K,N] -> bf16 transposed [N,K] (weights; one-time)
// ---------------------------------------------------------------------------
__global__ __launch_bounds__(256) void cvt_transpose_bf16(const float* __restrict__ in,
                                                          u16* __restrict__ out,
                                                          int K, int N) {
  const int idx = blockIdx.x * 256 + threadIdx.x;   // idx = k*N + n
  if (idx >= K * N) return;
  const int k = idx / N, n = idx - k * N;
  out[(size_t)n * K + k] = f2bf(in[idx]);
}

// ---------------------------------------------------------------------------
// LayerNorm over C=768, one row per 256-thread block, bf16 output
// ---------------------------------------------------------------------------
__global__ __launch_bounds__(256) void ln_bf16_kernel(const float* __restrict__ x,
                                                      const float* __restrict__ w,
                                                      const float* __restrict__ b,
                                                      u16* __restrict__ out) {
  const int row = blockIdx.x;
  const int tid = threadIdx.x;
  const float* xr = x + (size_t)row * EMBED;
  const float v0 = xr[tid], v1 = xr[tid + 256], v2 = xr[tid + 512];
  float s = v0 + v1 + v2;
  float s2 = v0 * v0 + v1 * v1 + v2 * v2;
#pragma unroll
  for (int off = 1; off < 32; off <<= 1) {
    s  += __shfl_xor(s,  off, 32);
    s2 += __shfl_xor(s2, off, 32);
  }
  __shared__ float rs[8], rs2[8];
  const int wid = tid >> 5, lane = tid & 31;
  if (lane == 0) { rs[wid] = s; rs2[wid] = s2; }
  __syncthreads();
  float fs = 0.f, fs2 = 0.f;
#pragma unroll
  for (int i = 0; i < 8; ++i) { fs += rs[i]; fs2 += rs2[i]; }
  const float mu   = fs * (1.0f / EMBED);
  const float var  = fs2 * (1.0f / EMBED) - mu * mu;
  const float rstd = rsqrtf(var + 1e-5f);
  u16* orow = out + (size_t)row * EMBED;
  orow[tid]       = f2bf((v0 - mu) * rstd * w[tid]       + b[tid]);
  orow[tid + 256] = f2bf((v1 - mu) * rstd * w[tid + 256] + b[tid + 256]);
  orow[tid + 512] = f2bf((v2 - mu) * rstd * w[tid + 512] + b[tid + 512]);
}

// ---------------------------------------------------------------------------
// bf16 GEMM: C[M,N] = A[M,K] * Bt[N,K]^T, 128x128 tile, K-step 32.
// MODE 0: +bias -> fp32   MODE 1: +bias+residual -> fp32
// MODE 2: +bias+GELU -> bf16
// Tile staging: TDM double-buffered (wave 0 issues, s_wait_tensorcnt), or
// synchronous vector copies as fallback.
// ---------------------------------------------------------------------------
template <int MODE>
__global__ __launch_bounds__(256) void gemm_bf16_wmma(
    const u16* __restrict__ A, const u16* __restrict__ Bt,
    const float* __restrict__ bias, const float* __restrict__ resid,
    float* __restrict__ outF, u16* __restrict__ outB,
    int M, int N, int K) {
  __shared__ __align__(16) u16 sA[2][128 * 40];
  __shared__ __align__(16) u16 sB[2][128 * 40];

  const int tid  = threadIdx.x;
  const int lane = tid & 31, wid = tid >> 5;
  const int wm = wid & 3, wn = wid >> 2;      // 4 waves along M, 2 along N
  const int m0 = blockIdx.y * 128, n0 = blockIdx.x * 128;

  v8f acc[2][4];
#pragma unroll
  for (int i = 0; i < 2; ++i)
#pragma unroll
    for (int j = 0; j < 4; ++j) acc[i][j] = v8f_zero();

  const int rsel  = lane & 15;
  const int abase = (lane < 16) ? 0 : 8;        // A frag K-base per ISA layout
  const int bbase = (lane < 16) ? 0 : 8;        // B frag K-base per ISA layout
  const int nk = K >> 5;

#ifdef USE_TDM
  if (wid == 0) {
    tdm_load_tile(&A [(size_t)m0 * K], (u32)(size_t)&sA[0][0], K, M - m0);
    tdm_load_tile(&Bt[(size_t)n0 * K], (u32)(size_t)&sB[0][0], K, N - n0);
  }
#else
  const int srow = tid >> 1, shalf = tid & 1;   // 128 rows x 2 halves
  const u16* gA = &A [(size_t)(m0 + srow) * K + shalf * 16];
  const u16* gB = &Bt[(size_t)(n0 + srow) * K + shalf * 16];
  {
    uint4* la = reinterpret_cast<uint4*>(&sA[0][srow * 40 + shalf * 16]);
    uint4* lb = reinterpret_cast<uint4*>(&sB[0][srow * 40 + shalf * 16]);
    la[0] = *reinterpret_cast<const uint4*>(gA);
    la[1] = *reinterpret_cast<const uint4*>(gA + 8);
    lb[0] = *reinterpret_cast<const uint4*>(gB);
    lb[1] = *reinterpret_cast<const uint4*>(gB + 8);
  }
#endif

  for (int kb = 0; kb < nk; ++kb) {
    const int cur = kb & 1;
#ifdef USE_TDM
    if (wid == 0) {
      if (kb + 1 < nk) {
        tdm_load_tile(&A [(size_t)m0 * K + (kb + 1) * 32],
                      (u32)(size_t)&sA[cur ^ 1][0], K, M - m0);
        tdm_load_tile(&Bt[(size_t)n0 * K + (kb + 1) * 32],
                      (u32)(size_t)&sB[cur ^ 1][0], K, N - n0);
        __builtin_amdgcn_s_wait_tensorcnt(2);   // current tile's 2 ops done
      } else {
        __builtin_amdgcn_s_wait_tensorcnt(0);
      }
    }
    __syncthreads();
#else
    __syncthreads();
#endif

    Frag af[2], bf[4];
#pragma unroll
    for (int i = 0; i < 2; ++i) {
      const u16* p = &sA[cur][(wm * 32 + i * 16 + rsel) * 40 + abase];
      af[i].q[0] = *reinterpret_cast<const uint4*>(p);
      af[i].q[1] = *reinterpret_cast<const uint4*>(p + 16);
    }
#pragma unroll
    for (int j = 0; j < 4; ++j) {
      const u16* p = &sB[cur][(wn * 64 + j * 16 + rsel) * 40 + bbase];
      bf[j].q[0] = *reinterpret_cast<const uint4*>(p);
      bf[j].q[1] = *reinterpret_cast<const uint4*>(p + 16);
    }
#pragma unroll
    for (int i = 0; i < 2; ++i)
#pragma unroll
      for (int j = 0; j < 4; ++j)
        acc[i][j] = __builtin_amdgcn_wmma_f32_16x16x32_bf16(
            false, af[i].v, false, bf[j].v, (short)0, acc[i][j], false, false);
    __syncthreads();

#ifndef USE_TDM
    if (kb + 1 < nk) {
      const int nxt = cur ^ 1;
      const int k0 = (kb + 1) * 32;
      uint4* la = reinterpret_cast<uint4*>(&sA[nxt][srow * 40 + shalf * 16]);
      uint4* lb = reinterpret_cast<uint4*>(&sB[nxt][srow * 40 + shalf * 16]);
      la[0] = *reinterpret_cast<const uint4*>(gA + k0);
      la[1] = *reinterpret_cast<const uint4*>(gA + k0 + 8);
      lb[0] = *reinterpret_cast<const uint4*>(gB + k0);
      lb[1] = *reinterpret_cast<const uint4*>(gB + k0 + 8);
    }
#endif
  }

  // ---- epilogue (compile-time mode, straight-line) ----
  const int rowoff = (lane >> 4) * 8;
#pragma unroll
  for (int i = 0; i < 2; ++i) {
#pragma unroll
    for (int j = 0; j < 4; ++j) {
      const int col = n0 + wn * 64 + j * 16 + rsel;
      const float bv = bias[col];
      const int row0 = m0 + wm * 32 + i * 16 + rowoff;
#pragma unroll
      for (int e = 0; e < 8; ++e) {
        const size_t o = (size_t)(row0 + e) * N + col;
        float v = acc[i][j][e] + bv;
        if constexpr (MODE == 2) {
          v = 0.5f * v * (1.0f + erff(v * 0.70710678118f));
          outB[o] = f2bf(v);
        } else if constexpr (MODE == 1) {
          outF[o] = v + resid[o];
        } else {
          outF[o] = v;
        }
      }
    }
  }
}

// ---------------------------------------------------------------------------
// Pack fp32 qkv [8192,2304] -> bf16 q/k [b,h,n,d] and v-transposed [b,h,d,n].
// 1/sqrt(HDIM) folded into q.
// ---------------------------------------------------------------------------
__global__ __launch_bounds__(256) void pack_qkv_kernel(const float* __restrict__ qkv,
                                                       u16* __restrict__ q,
                                                       u16* __restrict__ k,
                                                       u16* __restrict__ vt) {
  const int idx = blockIdx.x * 256 + threadIdx.x;  // ((b*H+h)*SEQ+n)*HDIM+d
  const int d  = idx & 63;
  const int n  = (idx >> 6) & 1023;
  const int bh = idx >> 16;                        // b*HEADS + h
  const int h  = bh % HEADS;
  const int bb = bh / HEADS;
  const size_t tok = (size_t)(bb * SEQ + n) * QKVN;
  const int col = h * HDIM + d;
  q[idx] = f2bf(qkv[tok + col] * 0.125f);
  k[idx] = f2bf(qkv[tok + EMBED + col]);
  vt[((size_t)bh * HDIM + d) * SEQ + n] = f2bf(qkv[tok + 2 * EMBED + col]);
}

// ---------------------------------------------------------------------------
// Flash attention: one wave owns 16 query rows; 32-key steps; all matmuls on
// WMMA bf16. Output written as bf16 [B,N,H*D] ready for the proj GEMM.
// ---------------------------------------------------------------------------
__global__ __launch_bounds__(256) void attn_flash_kernel(const u16* __restrict__ qb,
                                                         const u16* __restrict__ kb,
                                                         const u16* __restrict__ vt,
                                                         u16* __restrict__ outb) {
  __shared__ __align__(16) u16 sP[8][16 * 32];   // per-wave P scratch (C->A layout)
  const int tid = threadIdx.x, lane = tid & 31, wid = tid >> 5;
  const int bh = blockIdx.x >> 3, rb = blockIdx.x & 7;
  const int q0 = rb * 128 + wid * 16;
  const u16* Q  = qb + (size_t)bh * SEQ * HDIM;
  const u16* Kp = kb + (size_t)bh * SEQ * HDIM;
  const u16* Vt = vt + (size_t)bh * HDIM * SEQ;
  u16* myP = &sP[wid][0];

  const int rsel  = lane & 15;
  const int abase = (lane < 16) ? 0 : 8;
  const int bbase = (lane < 16) ? 0 : 8;

  Frag qf[2];
#pragma unroll
  for (int c = 0; c < 2; ++c) {
    const u16* p = &Q[(size_t)(q0 + rsel) * HDIM + c * 32 + abase];
    qf[c].q[0] = *reinterpret_cast<const uint4*>(p);
    qf[c].q[1] = *reinterpret_cast<const uint4*>(p + 16);
  }

  v8f o[4];
  float mrow[8], lrow[8];
#pragma unroll
  for (int j = 0; j < 4; ++j) o[j] = v8f_zero();
#pragma unroll
  for (int r = 0; r < 8; ++r) { mrow[r] = -3.0e38f; lrow[r] = 0.0f; }

  for (int kt = 0; kt < SEQ; kt += 32) {
    // ---- S = (Q * 1/8) . K^T for 32 keys (two 16-key tiles) ----
    v8f S[2];
#pragma unroll
    for (int s = 0; s < 2; ++s) {
      const u16* kr = &Kp[(size_t)(kt + s * 16 + rsel) * HDIM];
      Frag kf0, kf1;
      kf0.q[0] = *reinterpret_cast<const uint4*>(kr + bbase);
      kf0.q[1] = *reinterpret_cast<const uint4*>(kr + bbase + 16);
      kf1.q[0] = *reinterpret_cast<const uint4*>(kr + 32 + bbase);
      kf1.q[1] = *reinterpret_cast<const uint4*>(kr + 32 + bbase + 16);
      v8f z = v8f_zero();
      z = __builtin_amdgcn_wmma_f32_16x16x32_bf16(false, qf[0].v, false, kf0.v,
                                                  (short)0, z, false, false);
      S[s] = __builtin_amdgcn_wmma_f32_16x16x32_bf16(false, qf[1].v, false, kf1.v,
                                                     (short)0, z, false, false);
    }
    // ---- online softmax update (per row, cross-lane within 16-lane halves) ----
#pragma unroll
    for (int r = 0; r < 8; ++r) {
      float mx = fmaxf(S[0][r], S[1][r]);
#pragma unroll
      for (int off = 1; off < 16; off <<= 1) mx = fmaxf(mx, __shfl_xor(mx, off, 32));
      const float mnew  = fmaxf(mrow[r], mx);
      const float alpha = __expf(mrow[r] - mnew);
      mrow[r] = mnew;
      const float p0 = __expf(S[0][r] - mnew);
      const float p1 = __expf(S[1][r] - mnew);
      float rsum = p0 + p1;
#pragma unroll
      for (int off = 1; off < 16; off <<= 1) rsum += __shfl_xor(rsum, off, 32);
      lrow[r] = lrow[r] * alpha + rsum;
      const int prow = r + ((lane >> 4) << 3);
      myP[prow * 32 + rsel]      = f2bf(p0);
      myP[prow * 32 + 16 + rsel] = f2bf(p1);
#pragma unroll
      for (int j = 0; j < 4; ++j) o[j][r] *= alpha;
    }
    asm volatile("s_wait_dscnt 0" ::: "memory");   // wave-internal LDS transpose
    Frag pf;
    {
      const u16* p = &myP[rsel * 32 + abase];
      pf.q[0] = *reinterpret_cast<const uint4*>(p);
      pf.q[1] = *reinterpret_cast<const uint4*>(p + 16);
    }
    // ---- O += P . V (V stored d-major so B-frags are contiguous) ----
#pragma unroll
    for (int j = 0; j < 4; ++j) {
      const u16* vr = &Vt[(size_t)(j * 16 + rsel) * SEQ + kt + bbase];
      Frag vf;
      vf.q[0] = *reinterpret_cast<const uint4*>(vr);
      vf.q[1] = *reinterpret_cast<const uint4*>(vr + 16);
      o[j] = __builtin_amdgcn_wmma_f32_16x16x32_bf16(false, pf.v, false, vf.v,
                                                     (short)0, o[j], false, false);
    }
  }

  // ---- normalize and write bf16 [B,N,H*D] ----
  const int bb = bh / HEADS, h = bh % HEADS;
  float linv[8];
#pragma unroll
  for (int r = 0; r < 8; ++r) linv[r] = 1.0f / lrow[r];
#pragma unroll
  for (int j = 0; j < 4; ++j) {
    const int dcol = h * HDIM + j * 16 + rsel;
#pragma unroll
    for (int r = 0; r < 8; ++r) {
      const int row = q0 + r + ((lane >> 4) << 3);
      outb[(size_t)(bb * SEQ + row) * EMBED + dcol] = f2bf(o[j][r] * linv[r]);
    }
  }
}

// ---------------------------------------------------------------------------
// Host-side launch sequence
// ---------------------------------------------------------------------------
extern "C" void kernel_launch(void* const* d_in, const int* in_sizes, int n_in,
                              void* d_out, int out_size, void* d_ws, size_t ws_size,
                              hipStream_t stream) {
  (void)in_sizes; (void)n_in; (void)out_size; (void)ws_size;
  const float* x     = (const float*)d_in[0];
  const float* ln1w  = (const float*)d_in[1];
  const float* ln1b  = (const float*)d_in[2];
  const float* wqkv  = (const float*)d_in[3];
  const float* bqkv  = (const float*)d_in[4];
  const float* wproj = (const float*)d_in[5];
  const float* bproj = (const float*)d_in[6];
  const float* ln2w  = (const float*)d_in[7];
  const float* ln2b  = (const float*)d_in[8];
  const float* wfc1  = (const float*)d_in[9];
  const float* bfc1  = (const float*)d_in[10];
  const float* wfc2  = (const float*)d_in[11];
  const float* bfc2  = (const float*)d_in[12];
  float* out = (float*)d_out;

  char* ws = (char*)d_ws;
  size_t off = 0;
  auto alloc = [&](size_t bytes) -> char* {
    char* p = ws + off;
    off = (off + bytes + 255) & ~(size_t)255;
    return p;
  };

  u16*   hb     = (u16*)  alloc((size_t)TOKENS * EMBED  * 2);
  u16*   wqkvt  = (u16*)  alloc((size_t)EMBED  * QKVN   * 2);   // [2304][768]
  u16*   wprojt = (u16*)  alloc((size_t)EMBED  * EMBED  * 2);   // [768][768]
  u16*   wfc1t  = (u16*)  alloc((size_t)EMBED  * HIDDEN * 2);   // [3072][768]
  u16*   wfc2t  = (u16*)  alloc((size_t)HIDDEN * EMBED  * 2);   // [768][3072]
  float* qkvf   = (float*)alloc((size_t)TOKENS * QKVN   * 4);
  u16*   qbuf   = (u16*)  alloc((size_t)TOKENS * EMBED  * 2);
  u16*   kbuf   = (u16*)  alloc((size_t)TOKENS * EMBED  * 2);
  u16*   vtbuf  = (u16*)  alloc((size_t)TOKENS * EMBED  * 2);
  u16*   attnb  = (u16*)  alloc((size_t)TOKENS * EMBED  * 2);
  float* x1     = (float*)alloc((size_t)TOKENS * EMBED  * 4);
  u16*   h2b    = (u16*)  alloc((size_t)TOKENS * EMBED  * 2);
  u16*   gbuf   = (u16*)  alloc((size_t)TOKENS * HIDDEN * 2);

  // weights fp32 [K,N] -> bf16 transposed [N,K]
  cvt_transpose_bf16<<<(EMBED * QKVN + 255) / 256, 256, 0, stream>>>(wqkv, wqkvt, EMBED, QKVN);
  cvt_transpose_bf16<<<(EMBED * EMBED + 255) / 256, 256, 0, stream>>>(wproj, wprojt, EMBED, EMBED);
  cvt_transpose_bf16<<<(EMBED * HIDDEN + 255) / 256, 256, 0, stream>>>(wfc1, wfc1t, EMBED, HIDDEN);
  cvt_transpose_bf16<<<(HIDDEN * EMBED + 255) / 256, 256, 0, stream>>>(wfc2, wfc2t, HIDDEN, EMBED);

  // LN1 -> bf16
  ln_bf16_kernel<<<TOKENS, 256, 0, stream>>>(x, ln1w, ln1b, hb);

  // QKV GEMM: [8192,768] x [768,2304] + bias -> fp32
  gemm_bf16_wmma<0><<<dim3(QKVN / 128, TOKENS / 128), 256, 0, stream>>>(
      hb, wqkvt, bqkv, nullptr, qkvf, nullptr, TOKENS, QKVN, EMBED);

  // pack q/k/v (scale folded into q; v transposed d-major)
  pack_qkv_kernel<<<(TOKENS * EMBED) / 256, 256, 0, stream>>>(qkvf, qbuf, kbuf, vtbuf);

  // flash attention: 96 (b,h) pairs x 8 row-blocks
  attn_flash_kernel<<<BATCH * HEADS * 8, 256, 0, stream>>>(qbuf, kbuf, vtbuf, attnb);

  // proj GEMM + residual(x) -> x1 (fp32)
  gemm_bf16_wmma<1><<<dim3(EMBED / 128, TOKENS / 128), 256, 0, stream>>>(
      attnb, wprojt, bproj, x, x1, nullptr, TOKENS, EMBED, EMBED);

  // LN2 -> bf16
  ln_bf16_kernel<<<TOKENS, 256, 0, stream>>>(x1, ln2w, ln2b, h2b);

  // FC1 GEMM + bias + exact GELU -> bf16
  gemm_bf16_wmma<2><<<dim3(HIDDEN / 128, TOKENS / 128), 256, 0, stream>>>(
      h2b, wfc1t, bfc1, nullptr, nullptr, gbuf, TOKENS, HIDDEN, EMBED);

  // FC2 GEMM + bias + residual(x1) -> out (fp32)
  gemm_bf16_wmma<1><<<dim3(EMBED / 128, TOKENS / 128), 256, 0, stream>>>(
      gbuf, wfc2t, bfc2, x1, out, nullptr, TOKENS, EMBED, HIDDEN);
}
